// EuclideanRiemmanianAtt_90220083020131
// MI455X (gfx1250) — matrix operand order repaired and checked by reading, not compiled, not run on hardware
//
#include <hip/hip_runtime.h>
#include <hip/hip_bf16.h>

// ---------------------------------------------------------------------------
// CDNA5 (gfx1250) WMMA implementation of Euclidean+Riemannian attention.
// All GEMMs: v_wmma_f32_16x16x32_bf16 with hi/lo bf16 split of fp32 operands
// (bf16x3 -> ~fp32-class accuracy). Splits are precomputed once per tensor so
// GEMM inner loops are pure b128 loads + WMMA.
// ---------------------------------------------------------------------------

typedef __attribute__((ext_vector_type(16))) __bf16 v16bf;
typedef __attribute__((ext_vector_type(8)))  __bf16 v8bf;
typedef __attribute__((ext_vector_type(8)))  float  v8f;

#define BATCH 4
#define NSEQ  1024
#define CDIM  768
#define HEADS 12
#define DHEAD 64

__device__ __forceinline__ __bf16 f2bf(float x) {
    union { float f; unsigned u; } v; v.f = x;
    unsigned r = (v.u + 0x7fffu + ((v.u >> 16) & 1u)) >> 16;   // RNE
    union { unsigned short u; __bf16 b; } o; o.u = (unsigned short)r;
    return o.b;
}
__device__ __forceinline__ float bf2f(__bf16 b) {
    union { unsigned short u; __bf16 b; } i; i.b = b;
    union { unsigned u; float f; } o; o.u = ((unsigned)i.u) << 16;
    return o.f;
}
__device__ __forceinline__ void split_store(float v, __bf16* ph, __bf16* pl, size_t idx) {
    __bf16 h = f2bf(v);
    ph[idx] = h;
    pl[idx] = f2bf(v - bf2f(h));
}

// A fragment (16x32 MxK) from pre-split bf16 row-major [M][K] (ld in elements).
// lanes 0-15: M=lane, K = k0+0..7 , k0+16..23 ; lanes 16-31: M=lane-16, +8.
__device__ __forceinline__ v16bf load_a_bf(const __bf16* base, int ld, int row0, int k0) {
    int lane = threadIdx.x & 31;
    int half = lane >> 4;
    const __bf16* p = base + (size_t)(row0 + (lane & 15)) * ld + k0 + half * 8;
    v8bf a0 = *(const v8bf*)(p);
    v8bf a1 = *(const v8bf*)(p + 16);
    return __builtin_shufflevector(a0, a1, 0,1,2,3,4,5,6,7,8,9,10,11,12,13,14,15);
}

// B fragment (32x16 KxN) from pre-split bf16 B^T row-major [N][K].
// lanes 0-15: col=lane, K = k0+0..7 , k0+16..23 ; lanes 16-31: col=lane-16, +8.
__device__ __forceinline__ v16bf load_bt_bf(const __bf16* base, int ld, int col0, int k0) {
    int lane = threadIdx.x & 31;
    int half = lane >> 4;
    const __bf16* p = base + (size_t)(col0 + (lane & 15)) * ld + k0 + half * 8;
    v8bf b0 = *(const v8bf*)(p);
    v8bf b1 = *(const v8bf*)(p + 16);
    return __builtin_shufflevector(b0, b1, 0,1,2,3,4,5,6,7,8,9,10,11,12,13,14,15);
}

// bf16x3: D += A*B, dropping only the lo*lo term (~2^-16 relative error).
__device__ __forceinline__ v8f mm3(v16bf ah, v16bf al, v16bf bh, v16bf bl, v8f c) {
    c = __builtin_amdgcn_wmma_f32_16x16x32_bf16(false, ah, false, bh, (short)0, c, false, false);
    c = __builtin_amdgcn_wmma_f32_16x16x32_bf16(false, al, false, bh, (short)0, c, false, false);
    c = __builtin_amdgcn_wmma_f32_16x16x32_bf16(false, ah, false, bl, (short)0, c, false, false);
    return c;
}

// ---------------------------------------------------------------------------
// Prep: elementwise split (x) and transpose+split (weights)
// ---------------------------------------------------------------------------
__global__ void split_kernel(const float* __restrict__ in, __bf16* __restrict__ oh,
                             __bf16* __restrict__ ol, int n) {
    int idx = blockIdx.x * 256 + threadIdx.x;
    if (idx < n) split_store(in[idx], oh, ol, idx);
}

__global__ void transpose_split_kernel(const float* __restrict__ in,
                                       __bf16* __restrict__ oh, __bf16* __restrict__ ol,
                                       int rows, int cols) {
    int idx = blockIdx.x * 256 + threadIdx.x;
    if (idx < rows * cols) {
        int r = idx / cols, c = idx % cols;
        split_store(in[idx], oh, ol, (size_t)c * rows + r);
    }
}

// ---------------------------------------------------------------------------
// Kernel 1: QKV projection  [4096x768] x [768x2304] + bias.
// q,k -> fp32 head-major [B,H,N,D] (centering still needs fp32);
// v   -> transposed hi/lo bf16 [B,H,D,N].
// grid (18, 256), block 256 (8 waves).
// ---------------------------------------------------------------------------
__global__ __launch_bounds__(256) void qkv_kernel(
        const __bf16* __restrict__ xh, const __bf16* __restrict__ xl,
        const __bf16* __restrict__ wh, const __bf16* __restrict__ wl,
        const float* __restrict__ bqkv,
        float* __restrict__ qb, float* __restrict__ kb,
        __bf16* __restrict__ vTh, __bf16* __restrict__ vTl) {
    int wave = threadIdx.x >> 5;
    int n0 = blockIdx.x * 128 + wave * 16;   // output col 0..2303
    int m0 = blockIdx.y * 16;                // row 0..4095
    v8f acc = {};
    for (int k0 = 0; k0 < CDIM; k0 += 32) {
        v16bf ah = load_a_bf(xh, CDIM, m0, k0);
        v16bf al = load_a_bf(xl, CDIM, m0, k0);
        v16bf bh = load_bt_bf(wh, CDIM, n0, k0);
        v16bf bl = load_bt_bf(wl, CDIM, n0, k0);
        acc = mm3(ah, al, bh, bl, acc);
    }
    int lane = threadIdx.x & 31, half = lane >> 4;
    int col = n0 + (lane & 15);
    int which = col / CDIM;        // 0=q 1=k 2=v
    int r = col % CDIM;
    int h = r >> 6, d = r & 63;
    float bias = bqkv[col];
#pragma unroll
    for (int j = 0; j < 8; ++j) {
        int row = m0 + j + 8 * half;
        int b_ = row >> 10, n = row & 1023;
        float val = acc[j] + bias;
        size_t bh_ = (size_t)(b_ * HEADS + h);
        if (which == 0)      qb[(bh_ * NSEQ + n) * DHEAD + d] = val;
        else if (which == 1) kb[(bh_ * NSEQ + n) * DHEAD + d] = val;
        else                 split_store(val, vTh, vTl, (bh_ * DHEAD + d) * NSEQ + n);
    }
}

// ---------------------------------------------------------------------------
// Kernel 2: centering + split. Emits q,k and centered qc,kc as hi/lo bf16.
// grid (B*H*N, 2), block 64, LDS tree reduction for the mean over D=64.
// ---------------------------------------------------------------------------
__global__ void center_kernel(const float* __restrict__ q, const float* __restrict__ k,
                              __bf16* __restrict__ qh, __bf16* __restrict__ ql,
                              __bf16* __restrict__ qch, __bf16* __restrict__ qcl,
                              __bf16* __restrict__ kh, __bf16* __restrict__ kl,
                              __bf16* __restrict__ kch, __bf16* __restrict__ kcl) {
    __shared__ float s[64];
    int row = blockIdx.x;
    const float* src;
    __bf16 *dh, *dl, *ch, *cl;
    if (blockIdx.y == 0) { src = q; dh = qh; dl = ql; ch = qch; cl = qcl; }
    else                 { src = k; dh = kh; dl = kl; ch = kch; cl = kcl; }
    int t = threadIdx.x;
    size_t idx = (size_t)row * DHEAD + t;
    float v = src[idx];
    s[t] = v;
    __syncthreads();
    for (int st = 32; st > 0; st >>= 1) {
        if (t < st) s[t] += s[t + st];
        __syncthreads();
    }
    float mean = s[0] * (1.0f / 64.0f);
    split_store(v, dh, dl, idx);
    split_store(v - mean, ch, cl, idx);
}

// ---------------------------------------------------------------------------
// Kernel 3: fused scores + channel mix. Wave owns a 16(n)x16(m) tile, loops
// all 12 heads, keeps 12 mixed accumulators. 144 v_wmma per wave. Pre-softmax
// result goes straight into d_out's attn_ region.
// grid (8, 64, 4), block 256.
// ---------------------------------------------------------------------------
__global__ __launch_bounds__(256) void scores_kernel(
        const __bf16* __restrict__ qh, const __bf16* __restrict__ ql,
        const __bf16* __restrict__ qch, const __bf16* __restrict__ qcl,
        const __bf16* __restrict__ kh, const __bf16* __restrict__ kl,
        const __bf16* __restrict__ kch, const __bf16* __restrict__ kcl,
        const float* __restrict__ scale_p, const float* __restrict__ rscale_p,
        const float* __restrict__ Wconv, const float* __restrict__ bconv,
        float* __restrict__ attn) {
    __shared__ float sWc[HEADS * 2 * HEADS];
    __shared__ float sBc[HEADS];
    int t = threadIdx.x;
    for (int i = t; i < HEADS * 2 * HEADS; i += 256) sWc[i] = Wconv[i];
    if (t < HEADS) sBc[t] = bconv[t];
    __syncthreads();

    int wave = t >> 5, lane = t & 31, half = lane >> 4;
    int b = blockIdx.z;
    int n0 = blockIdx.y * 16;
    int m0 = blockIdx.x * 128 + wave * 16;
    float scale = scale_p[0], rscale = rscale_p[0];

    v8f acc[HEADS];
#pragma unroll
    for (int o = 0; o < HEADS; ++o) { v8f z = {}; acc[o] = z; }

    for (int h = 0; h < HEADS; ++h) {
        size_t off = (size_t)(b * HEADS + h) * NSEQ * DHEAD;
        v8f se = {}, sr = {};
#pragma unroll
        for (int k0 = 0; k0 < DHEAD; k0 += 32) {
            se = mm3(load_a_bf(qh + off, DHEAD, n0, k0),  load_a_bf(ql + off, DHEAD, n0, k0),
                     load_bt_bf(kh + off, DHEAD, m0, k0), load_bt_bf(kl + off, DHEAD, m0, k0), se);
            sr = mm3(load_a_bf(qch + off, DHEAD, n0, k0),  load_a_bf(qcl + off, DHEAD, n0, k0),
                     load_bt_bf(kch + off, DHEAD, m0, k0), load_bt_bf(kcl + off, DHEAD, m0, k0), sr);
        }
#pragma unroll
        for (int j = 0; j < 8; ++j) {
            se[j] *= scale;
            float s = sr[j];
            sr[j] = s * s * rscale;
        }
#pragma unroll
        for (int o = 0; o < HEADS; ++o) {
            float we = sWc[o * 24 + h];
            float wr = sWc[o * 24 + HEADS + h];
#pragma unroll
            for (int j = 0; j < 8; ++j) acc[o][j] += we * se[j] + wr * sr[j];
        }
    }

#pragma unroll
    for (int o = 0; o < HEADS; ++o) {
        float bias = sBc[o];
        size_t base = (size_t)(b * HEADS + o) * NSEQ * NSEQ;
#pragma unroll
        for (int j = 0; j < 8; ++j) {
            int n = n0 + j + 8 * half;
            attn[base + (size_t)n * NSEQ + m0 + (lane & 15)] = acc[o][j] + bias;
        }
    }
}

// ---------------------------------------------------------------------------
// Kernel 4: in-place softmax over last dim; also emits probabilities as hi/lo
// bf16 (A-operand of attn@v). grid (B*H*N), block 256.
// ---------------------------------------------------------------------------
__global__ __launch_bounds__(256) void softmax_kernel(float* __restrict__ a,
                                                      __bf16* __restrict__ ph,
                                                      __bf16* __restrict__ pl) {
    __shared__ float red[256];
    int row = blockIdx.x;
    float* p = a + (size_t)row * NSEQ;
    int t = threadIdx.x;
    float vmax = -3.0e38f;
    for (int i = t; i < NSEQ; i += 256) vmax = fmaxf(vmax, p[i]);
    red[t] = vmax;
    __syncthreads();
    for (int s = 128; s > 0; s >>= 1) {
        if (t < s) red[t] = fmaxf(red[t], red[t + s]);
        __syncthreads();
    }
    vmax = red[0];
    __syncthreads();
    float sum = 0.0f;
    for (int i = t; i < NSEQ; i += 256) {
        float e = __expf(p[i] - vmax);
        p[i] = e;
        sum += e;
    }
    red[t] = sum;
    __syncthreads();
    for (int s = 128; s > 0; s >>= 1) {
        if (t < s) red[t] += red[t + s];
        __syncthreads();
    }
    float inv = 1.0f / red[0];
    size_t rbase = (size_t)row * NSEQ;
    for (int i = t; i < NSEQ; i += 256) {
        float pv = p[i] * inv;
        p[i] = pv;
        split_store(pv, ph, pl, rbase + i);
    }
}

// ---------------------------------------------------------------------------
// Kernel 5: out_h = attn_ @ v (per head, K=1024). 96 v_wmma per wave.
// grid (64, 48), block 128 (4 waves = 4 d-subtiles). Emits out_h hi/lo bf16.
// ---------------------------------------------------------------------------
__global__ __launch_bounds__(128) void av_kernel(
        const __bf16* __restrict__ ph, const __bf16* __restrict__ pl,
        const __bf16* __restrict__ vTh, const __bf16* __restrict__ vTl,
        __bf16* __restrict__ oh, __bf16* __restrict__ ol) {
    int wave = threadIdx.x >> 5;            // d-tile 0..3
    int bh = blockIdx.y;
    int n0 = blockIdx.x * 16;
    const __bf16* Ah = ph + (size_t)bh * NSEQ * NSEQ;
    const __bf16* Al = pl + (size_t)bh * NSEQ * NSEQ;
    const __bf16* Bh = vTh + (size_t)bh * DHEAD * NSEQ;
    const __bf16* Bl = vTl + (size_t)bh * DHEAD * NSEQ;
    v8f acc = {};
    for (int k0 = 0; k0 < NSEQ; k0 += 32) {
        acc = mm3(load_a_bf(Ah, NSEQ, n0, k0),        load_a_bf(Al, NSEQ, n0, k0),
                  load_bt_bf(Bh, NSEQ, wave * 16, k0), load_bt_bf(Bl, NSEQ, wave * 16, k0), acc);
    }
    int lane = threadIdx.x & 31, half = lane >> 4;
    int b_ = bh / HEADS, h = bh % HEADS;
#pragma unroll
    for (int j = 0; j < 8; ++j) {
        int n = n0 + j + 8 * half;
        size_t idx = (size_t)(b_ * NSEQ + n) * CDIM + h * DHEAD + wave * 16 + (lane & 15);
        split_store(acc[j], oh, ol, idx);
    }
}

// ---------------------------------------------------------------------------
// Kernel 6: output projection [4096x768] x [768x768] + bias -> d_out.
// grid (6, 256), block 256.
// ---------------------------------------------------------------------------
__global__ __launch_bounds__(256) void proj_kernel(
        const __bf16* __restrict__ oh, const __bf16* __restrict__ ol,
        const __bf16* __restrict__ wh, const __bf16* __restrict__ wl,
        const float* __restrict__ bproj, float* __restrict__ out) {
    int wave = threadIdx.x >> 5;
    int n0 = blockIdx.x * 128 + wave * 16;
    int m0 = blockIdx.y * 16;
    v8f acc = {};
    for (int k0 = 0; k0 < CDIM; k0 += 32) {
        acc = mm3(load_a_bf(oh, CDIM, m0, k0),  load_a_bf(ol, CDIM, m0, k0),
                  load_bt_bf(wh, CDIM, n0, k0), load_bt_bf(wl, CDIM, n0, k0), acc);
    }
    int lane = threadIdx.x & 31, half = lane >> 4;
    int col = n0 + (lane & 15);
    float bias = bproj[col];
#pragma unroll
    for (int j = 0; j < 8; ++j) {
        int row = m0 + j + 8 * half;
        out[(size_t)row * CDIM + col] = acc[j] + bias;
    }
}

// ---------------------------------------------------------------------------
extern "C" void kernel_launch(void* const* d_in, const int* in_sizes, int n_in,
                              void* d_out, int out_size, void* d_ws, size_t ws_size,
                              hipStream_t stream) {
    (void)in_sizes; (void)n_in; (void)out_size; (void)ws_size;
    const float* x      = (const float*)d_in[0];   // [B,N,C]
    const float* W_qkv  = (const float*)d_in[1];   // [C,3C]
    const float* b_qkv  = (const float*)d_in[2];   // [3C]
    const float* scale  = (const float*)d_in[3];   // [1]
    const float* rscale = (const float*)d_in[4];   // [1]
    const float* W_conv = (const float*)d_in[5];   // [H,2H]
    const float* b_conv = (const float*)d_in[6];   // [H]
    const float* W_proj = (const float*)d_in[7];   // [C,C]
    const float* b_proj = (const float*)d_in[8];   // [C]

    float* out  = (float*)d_out;                                   // [B,N,C]
    float* attn = (float*)d_out + (size_t)BATCH * NSEQ * CDIM;     // [B,H,N,N]

    const size_t SZ_HEAD = (size_t)BATCH * HEADS * NSEQ * DHEAD;   // 3,145,728
    const size_t SZ_X    = (size_t)BATCH * NSEQ * CDIM;            // 3,145,728
    const size_t SZ_ATT  = (size_t)BATCH * HEADS * NSEQ * NSEQ;    // 50,331,648

    // fp32 scratch
    float* wsf = (float*)d_ws;
    float* qb = wsf;  wsf += SZ_HEAD;
    float* kb = wsf;  wsf += SZ_HEAD;
    // bf16 scratch (16B-aligned: all sizes are multiples of 8)
    __bf16* wsb = (__bf16*)wsf;
    __bf16* xh  = wsb; wsb += SZ_X;
    __bf16* xl  = wsb; wsb += SZ_X;
    __bf16* Wqh = wsb; wsb += (size_t)CDIM * 3 * CDIM;
    __bf16* Wql = wsb; wsb += (size_t)CDIM * 3 * CDIM;
    __bf16* Wph = wsb; wsb += (size_t)CDIM * CDIM;
    __bf16* Wpl = wsb; wsb += (size_t)CDIM * CDIM;
    __bf16* qh  = wsb; wsb += SZ_HEAD;
    __bf16* ql  = wsb; wsb += SZ_HEAD;
    __bf16* qch = wsb; wsb += SZ_HEAD;
    __bf16* qcl = wsb; wsb += SZ_HEAD;
    __bf16* kh  = wsb; wsb += SZ_HEAD;
    __bf16* kl  = wsb; wsb += SZ_HEAD;
    __bf16* kch = wsb; wsb += SZ_HEAD;
    __bf16* kcl = wsb; wsb += SZ_HEAD;
    __bf16* vTh = wsb; wsb += SZ_HEAD;
    __bf16* vTl = wsb; wsb += SZ_HEAD;
    __bf16* ohh = wsb; wsb += SZ_X;
    __bf16* ohl = wsb; wsb += SZ_X;
    __bf16* ph  = wsb; wsb += SZ_ATT;
    __bf16* pl  = wsb; wsb += SZ_ATT;

    // 0) operand prep: split x, transpose+split weights
    split_kernel<<<(int)((SZ_X + 255) / 256), 256, 0, stream>>>(x, xh, xl, (int)SZ_X);
    transpose_split_kernel<<<(CDIM * 3 * CDIM + 255) / 256, 256, 0, stream>>>(
        W_qkv, Wqh, Wql, CDIM, 3 * CDIM);
    transpose_split_kernel<<<(CDIM * CDIM + 255) / 256, 256, 0, stream>>>(
        W_proj, Wph, Wpl, CDIM, CDIM);

    // 1) QKV projection (WMMA)
    qkv_kernel<<<dim3(3 * CDIM / 128, BATCH * NSEQ / 16), 256, 0, stream>>>(
        xh, xl, Wqh, Wql, b_qkv, qb, kb, vTh, vTl);

    // 2) centering + split
    center_kernel<<<dim3(BATCH * HEADS * NSEQ, 2), 64, 0, stream>>>(
        qb, kb, qh, ql, qch, qcl, kh, kl, kch, kcl);

    // 3) fused scores + channel mix (WMMA) -> pre-softmax attn region of d_out
    scores_kernel<<<dim3(NSEQ / 128, NSEQ / 16, BATCH), 256, 0, stream>>>(
        qh, ql, qch, qcl, kh, kl, kch, kcl, scale, rscale, W_conv, b_conv, attn);

    // 4) softmax in place + split probabilities
    softmax_kernel<<<BATCH * HEADS * NSEQ, 256, 0, stream>>>(attn, ph, pl);

    // 5) attn @ v (WMMA)
    av_kernel<<<dim3(NSEQ / 16, BATCH * HEADS), 128, 0, stream>>>(ph, pl, vTh, vTl, ohh, ohl);

    // 6) output projection (WMMA) -> d_out
    proj_kernel<<<dim3(CDIM / 128, BATCH * NSEQ / 16), 256, 0, stream>>>(
        ohh, ohl, Wph, Wpl, b_proj, out);
}
